// GNNLayer_19679540150749
// MI455X (gfx1250) — compile-verified, matrix operand order checked by reading, not run on hardware
//
#include <hip/hip_runtime.h>
#include <hip/hip_bf16.h>

// Problem constants (from reference):
#define B_   4
#define N_   4096
#define E_   65536
#define D_   256
#define ED_  64
#define CD_  128
#define H_   256
#define K1   (D_ + ED_ + CD_)   // 448  message MLP1 input
#define K2   H_                 // 256  message MLP2 / update MLP2 input
#define KU1  (D_ + H_ + CD_)    // 640  update MLP1 input
#define BDIM 256

typedef __bf16 bf16;
typedef __attribute__((ext_vector_type(16))) __bf16 v16bf;
typedef __attribute__((ext_vector_type(8)))  float  v8f;

union FragU { v16bf v; uint4 q[2]; };

// Build a 16x32 bf16 WMMA operand fragment (A or B) from row-major bf16
// storage with row stride `rowstride` (halves). Per ISA 7.12.2:
//   lanes 0-15 : row = lane,      K-halves 0..7  and 16..23
//   lanes 16-31: row = lane-16,   K-halves 8..15 and 24..31
// Two contiguous 16-byte loads per lane.
__device__ __forceinline__ v16bf ldfrag(const bf16* base, int rowstride,
                                        int row_base, int kbase, int lane) {
    const bf16* p = base + (size_t)(row_base + (lane & 15)) * rowstride
                         + kbase + ((lane & 16) ? 8 : 0);
    FragU f;
    f.q[0] = *(const uint4*)(p);
    f.q[1] = *(const uint4*)(p + 16);
    return f.v;
}

__device__ __forceinline__ float silu(float x) {
    return x / (1.0f + __expf(-x));
}

#define WMMA_BF16(C, A, Bv) \
    __builtin_amdgcn_wmma_f32_16x16x32_bf16(false, (A), false, (Bv), (short)0, (C), false, false)

// ---------------------------------------------------------------------------
// Prep kernels
// ---------------------------------------------------------------------------
__global__ __launch_bounds__(BDIM) void zero_kernel(float* __restrict__ p, int n) {
    int i = blockIdx.x * BDIM + threadIdx.x;
    if (i < n) p[i] = 0.0f;
}

// W[k][n] f32 -> WT[n][k] bf16  (WMMA-B-friendly: contiguous along K)
__global__ __launch_bounds__(BDIM) void convT_kernel(const float* __restrict__ W,
                                                     bf16* __restrict__ WT,
                                                     int K, int Nn) {
    int i = blockIdx.x * BDIM + threadIdx.x;
    if (i < K * Nn) {
        int k = i / Nn, n = i - k * Nn;
        WT[(size_t)n * K + k] = (bf16)W[i];
    }
}

// ---------------------------------------------------------------------------
// Message kernel: per 16-edge tile
//   msg = silu(silu([h_src|edge_attr|cond] @ Wm1 + bm1) @ Wm2 + bm2) * emask
//   agg[b, dst, :] += msg   (f32 global atomics)
// ---------------------------------------------------------------------------
__global__ __launch_bounds__(BDIM) void msg_kernel(
    const float* __restrict__ h, const int* __restrict__ edge_index,
    const float* __restrict__ edge_attr, const float* __restrict__ cond,
    const float* __restrict__ edge_mask,
    const bf16* __restrict__ Wm1T, const float* __restrict__ bm1,
    const bf16* __restrict__ Wm2T, const float* __restrict__ bm2,
    float* __restrict__ agg)
{
    __shared__ __align__(16) bf16 As[16 * K1];   // 14336 B
    __shared__ __align__(16) bf16 Hs[16 * H_];   //  8192 B
    __shared__ int   s_src[16];
    __shared__ int   s_dst[16];
    __shared__ float s_em[16];

    const int tid   = threadIdx.x;
    const int tile  = blockIdx.x;
    const int b     = blockIdx.y;
    const int ebase = tile * 16;

    if (tid < 16) {
        s_src[tid] = edge_index[ebase + tid];          // edge_index[0][e]
        s_dst[tid] = edge_index[E_ + ebase + tid];     // edge_index[1][e]
        s_em[tid]  = edge_mask[(size_t)b * E_ + ebase + tid];
    }
    __syncthreads();

    // Stage 16 x 448 concatenated input rows into LDS (f32 -> bf16).
    for (int i = tid; i < 16 * K1; i += BDIM) {
        int row = i / K1, col = i - row * K1;
        float v;
        if (col < D_)            v = h[((size_t)b * N_ + s_src[row]) * D_ + col];
        else if (col < D_ + ED_) v = edge_attr[((size_t)b * E_ + ebase + row) * ED_ + (col - D_)];
        else                     v = cond[(size_t)b * CD_ + (col - D_ - ED_)];
        As[i] = (bf16)v;
    }
    __syncthreads();

    const int wave = tid >> 5;
    const int lane = tid & 31;
    const int n0   = wave * 32;            // this wave owns cols [n0, n0+32)
    const int col0 = n0 + (lane & 15);
    const int col1 = n0 + 16 + (lane & 15);
    const int rb   = (lane & 16) ? 8 : 0;  // C-fragment row base

    // ---- GEMM1: 16x448 @ 448x32 -> 16x32 (two WMMA tiles per wave) ----
    v8f c0 = {}, c1 = {};
    for (int kc = 0; kc < K1; kc += 32) {
        __builtin_prefetch(Wm1T + (size_t)col0 * K1 + kc + 32, 0, 1);
        v16bf a  = ldfrag(As,   K1, 0,       kc, lane);
        v16bf b0 = ldfrag(Wm1T, K1, n0,      kc, lane);
        v16bf b1 = ldfrag(Wm1T, K1, n0 + 16, kc, lane);
        c0 = WMMA_BF16(c0, a, b0);
        c1 = WMMA_BF16(c1, a, b1);
    }
    {
        float bi0 = bm1[col0], bi1 = bm1[col1];
        #pragma unroll
        for (int r = 0; r < 8; ++r) {
            Hs[(rb + r) * H_ + col0] = (bf16)silu(c0[r] + bi0);
            Hs[(rb + r) * H_ + col1] = (bf16)silu(c1[r] + bi1);
        }
    }
    __syncthreads();

    // ---- GEMM2: 16x256 @ 256x32 -> 16x32 ----
    v8f d0 = {}, d1 = {};
    for (int kc = 0; kc < K2; kc += 32) {
        v16bf a  = ldfrag(Hs,   H_, 0,       kc, lane);
        v16bf b0 = ldfrag(Wm2T, K2, n0,      kc, lane);
        v16bf b1 = ldfrag(Wm2T, K2, n0 + 16, kc, lane);
        d0 = WMMA_BF16(d0, a, b0);
        d1 = WMMA_BF16(d1, a, b1);
    }
    {
        float bi0 = bm2[col0], bi1 = bm2[col1];
        #pragma unroll
        for (int r = 0; r < 8; ++r) {
            int row = rb + r;
            float em = s_em[row];
            size_t nb = ((size_t)b * N_ + s_dst[row]) * (size_t)H_;
            atomicAdd(&agg[nb + col0], silu(d0[r] + bi0) * em);
            atomicAdd(&agg[nb + col1], silu(d1[r] + bi1) * em);
        }
    }
}

// ---------------------------------------------------------------------------
// Update kernel: per 16-node tile
//   dh = silu([h|agg|cond] @ Wu1 + bu1) @ Wu2 + bu2 ; x = h + dh
//   out = LN(x)*gamma + beta, masked
// ---------------------------------------------------------------------------
__global__ __launch_bounds__(BDIM) void upd_kernel(
    const float* __restrict__ h, const float* __restrict__ cond,
    const unsigned char* __restrict__ mask,
    const float* __restrict__ agg,
    const bf16* __restrict__ Wu1T, const float* __restrict__ bu1,
    const bf16* __restrict__ Wu2T, const float* __restrict__ bu2,
    const float* __restrict__ gamma, const float* __restrict__ beta,
    float* __restrict__ out)
{
    __shared__ __align__(16) unsigned char smem[16 * KU1 * 2 + 16 * H_ * 2]; // 28672 B
    bf16*  As = (bf16*)smem;                      // 16 x 640 bf16
    bf16*  Hs = (bf16*)(smem + 16 * KU1 * 2);     // 16 x 256 bf16
    float* Xs = (float*)smem;                     // 16 x 256 f32 (aliases As,
                                                  // only used after all As reads)

    const int tid   = threadIdx.x;
    const int tile  = blockIdx.x;
    const int b     = blockIdx.y;
    const int nbase = tile * 16;

    // Stage 16 x 640 rows [h | agg | cond] into LDS as bf16.
    for (int i = tid; i < 16 * KU1; i += BDIM) {
        int row = i / KU1, col = i - row * KU1;
        size_t nrow = (size_t)b * N_ + nbase + row;
        float v;
        if (col < D_)            v = h[nrow * D_ + col];
        else if (col < D_ + H_)  v = agg[nrow * H_ + (col - D_)];
        else                     v = cond[(size_t)b * CD_ + (col - D_ - H_)];
        As[i] = (bf16)v;
    }
    __syncthreads();

    const int wave = tid >> 5;
    const int lane = tid & 31;
    const int n0   = wave * 32;
    const int col0 = n0 + (lane & 15);
    const int col1 = n0 + 16 + (lane & 15);
    const int rb   = (lane & 16) ? 8 : 0;

    // ---- GEMM1: 16x640 @ 640x32 ----
    v8f c0 = {}, c1 = {};
    for (int kc = 0; kc < KU1; kc += 32) {
        __builtin_prefetch(Wu1T + (size_t)col0 * KU1 + kc + 32, 0, 1);
        v16bf a  = ldfrag(As,   KU1, 0,       kc, lane);
        v16bf b0 = ldfrag(Wu1T, KU1, n0,      kc, lane);
        v16bf b1 = ldfrag(Wu1T, KU1, n0 + 16, kc, lane);
        c0 = WMMA_BF16(c0, a, b0);
        c1 = WMMA_BF16(c1, a, b1);
    }
    {
        float bi0 = bu1[col0], bi1 = bu1[col1];
        #pragma unroll
        for (int r = 0; r < 8; ++r) {
            Hs[(rb + r) * H_ + col0] = (bf16)silu(c0[r] + bi0);
            Hs[(rb + r) * H_ + col1] = (bf16)silu(c1[r] + bi1);
        }
    }
    __syncthreads();  // all As reads complete beyond this point -> Xs may alias As

    // ---- GEMM2: 16x256 @ 256x32 ----
    v8f d0 = {}, d1 = {};
    for (int kc = 0; kc < K2; kc += 32) {
        v16bf a  = ldfrag(Hs,   H_, 0,       kc, lane);
        v16bf b0 = ldfrag(Wu2T, K2, n0,      kc, lane);
        v16bf b1 = ldfrag(Wu2T, K2, n0 + 16, kc, lane);
        d0 = WMMA_BF16(d0, a, b0);
        d1 = WMMA_BF16(d1, a, b1);
    }
    {
        float bi0 = bu2[col0], bi1 = bu2[col1];
        #pragma unroll
        for (int r = 0; r < 8; ++r) {
            int row = rb + r;
            size_t nrow = (size_t)b * N_ + nbase + row;
            Xs[row * D_ + col0] = h[nrow * D_ + col0] + d0[r] + bi0;  // residual
            Xs[row * D_ + col1] = h[nrow * D_ + col1] + d1[r] + bi1;
        }
    }
    __syncthreads();

    // ---- LayerNorm over D=256 per row; wave32 shuffle reduction ----
    #pragma unroll
    for (int rr = 0; rr < 2; ++rr) {
        int row  = wave * 2 + rr;
        int node = nbase + row;
        float s = 0.0f, s2 = 0.0f;
        for (int j = lane; j < D_; j += 32) {
            float x = Xs[row * D_ + j];
            s += x; s2 += x * x;
        }
        #pragma unroll
        for (int off = 16; off >= 1; off >>= 1) {
            s  += __shfl_xor(s,  off, 32);
            s2 += __shfl_xor(s2, off, 32);
        }
        float mu  = s  * (1.0f / (float)D_);
        float var = s2 * (1.0f / (float)D_) - mu * mu;
        float rs  = rsqrtf(var + 1e-5f);
        int mk = (int)mask[(size_t)b * N_ + node];
        size_t obase = ((size_t)b * N_ + node) * (size_t)D_;
        for (int j = lane; j < D_; j += 32) {
            float x = Xs[row * D_ + j];
            float o = (x - mu) * rs * gamma[j] + beta[j];
            out[obase + j] = mk ? o : h[obase + j];
        }
    }
}

// ---------------------------------------------------------------------------
extern "C" void kernel_launch(void* const* d_in, const int* in_sizes, int n_in,
                              void* d_out, int out_size, void* d_ws, size_t ws_size,
                              hipStream_t stream)
{
    const float* h          = (const float*)d_in[0];
    const int*   edge_index = (const int*)d_in[1];
    const float* edge_attr  = (const float*)d_in[2];
    const float* cond       = (const float*)d_in[3];
    const unsigned char* mask = (const unsigned char*)d_in[4];
    const float* edge_mask  = (const float*)d_in[5];
    const float* Wm1 = (const float*)d_in[6];
    const float* bm1 = (const float*)d_in[7];
    const float* Wm2 = (const float*)d_in[8];
    const float* bm2 = (const float*)d_in[9];
    const float* Wu1 = (const float*)d_in[10];
    const float* bu1 = (const float*)d_in[11];
    const float* Wu2 = (const float*)d_in[12];
    const float* bu2 = (const float*)d_in[13];
    const float* gamma = (const float*)d_in[14];
    const float* beta  = (const float*)d_in[15];
    float* out = (float*)d_out;

    // Workspace layout: [agg f32 16MB][Wm1T][Wm2T][Wu1T][Wu2T] (bf16, ~0.8MB)
    const size_t aggN = (size_t)B_ * N_ * H_;                 // 4,194,304 floats
    float* agg = (float*)d_ws;
    bf16* Wm1T = (bf16*)((char*)d_ws + aggN * sizeof(float));
    bf16* Wm2T = Wm1T + (size_t)H_ * K1;
    bf16* Wu1T = Wm2T + (size_t)H_ * K2;
    bf16* Wu2T = Wu1T + (size_t)H_ * KU1;

    zero_kernel<<<(int)((aggN + BDIM - 1) / BDIM), BDIM, 0, stream>>>(agg, (int)aggN);
    convT_kernel<<<(K1 * H_ + BDIM - 1) / BDIM, BDIM, 0, stream>>>(Wm1, Wm1T, K1, H_);
    convT_kernel<<<(K2 * H_ + BDIM - 1) / BDIM, BDIM, 0, stream>>>(Wm2, Wm2T, K2, H_);
    convT_kernel<<<(KU1 * H_ + BDIM - 1) / BDIM, BDIM, 0, stream>>>(Wu1, Wu1T, KU1, H_);
    convT_kernel<<<(K2 * D_ + BDIM - 1) / BDIM, BDIM, 0, stream>>>(Wu2, Wu2T, K2, D_);

    dim3 gm(E_ / 16, B_);
    msg_kernel<<<gm, BDIM, 0, stream>>>(h, edge_index, edge_attr, cond, edge_mask,
                                        Wm1T, bm1, Wm2T, bm2, agg);
    dim3 gu(N_ / 16, B_);
    upd_kernel<<<gu, BDIM, 0, stream>>>(h, cond, mask, agg, Wu1T, bu1, Wu2T, bu2,
                                        gamma, beta, out);

    (void)in_sizes; (void)n_in; (void)out_size; (void)ws_size;
}
